// GraphVAE_49297634623704
// MI455X (gfx1250) — matrix operand order repaired and checked by reading, not compiled, not run on hardware
//
#include <hip/hip_runtime.h>
#include <hip/hip_bf16.h>
#include <math.h>

typedef __attribute__((ext_vector_type(16))) _Float16 v16h;
typedef __attribute__((ext_vector_type(8)))  _Float16 v8h;
typedef __attribute__((ext_vector_type(8)))  float    v8f;

__device__ __forceinline__ float elu1(float v) { return v > 0.f ? v : expm1f(v); }

__device__ __forceinline__ v16h cat8(v8h lo, v8h hi) {
    return __builtin_shufflevector(lo, hi, 0, 1, 2, 3, 4, 5, 6, 7,
                                           8, 9, 10, 11, 12, 13, 14, 15);
}

// ---------------------------------------------------------------------------
// SplineConv message + scatter. One thread per (edge, out-channel).
// agg must be pre-zeroed. x:[Nn,CIN], ei:[2,E], ea:[E,3], W:[125,CIN,COUT].
// ---------------------------------------------------------------------------
template<int CIN, int COUT>
__global__ void spline_msg_kernel(const float* __restrict__ x,
                                  const int*   __restrict__ ei,
                                  const float* __restrict__ ea,
                                  const float* __restrict__ W,
                                  float* __restrict__ agg,
                                  int E)
{
    long t = (long)blockIdx.x * blockDim.x + threadIdx.x;
    long e = t / COUT;                      // COUT is a power of two -> shifts
    int  co = (int)(t % COUT);
    if (e >= E) return;
    __builtin_prefetch(ei + e + 64, 0, 1);  // global_prefetch_b8
    int src = ei[e];
    int dst = ei[E + e];

    float fr[3]; int lo[3];
#pragma unroll
    for (int d = 0; d < 3; ++d) {
        float v = ea[e * 3 + d] * 4.0f;     // (KS-1) = 4
        float l = floorf(v);
        l = fminf(fmaxf(l, 0.f), 3.f);      // clip to [0, KS-2]
        fr[d] = v - l;
        lo[d] = (int)l;
    }
    float basis[8]; int kidx[8];
#pragma unroll
    for (int s = 0; s < 8; ++s) {
        float bs = 1.f; int k = 0; int strd = 1;
#pragma unroll
        for (int d = 0; d < 3; ++d) {
            int bit = (s >> d) & 1;
            bs *= bit ? fr[d] : (1.f - fr[d]);
            k  += (lo[d] + bit) * strd;
            strd *= 5;
        }
        basis[s] = bs; kidx[s] = k;
    }

    const float* xs = x + (size_t)src * CIN;
    float acc = 0.f;
    for (int ci = 0; ci < CIN; ++ci) {
        float xv = xs[ci];
#pragma unroll
        for (int s = 0; s < 8; ++s)
            acc += basis[s] * xv * W[((size_t)kidx[s] * CIN + ci) * COUT + co];
    }
    atomicAdd(&agg[(size_t)dst * COUT + co], acc);
}

// ---------------------------------------------------------------------------
// out[n,co] = ELU( agg[n,co] + x[n,:] @ root[:,co] + bias[co] )
// Also emits an f16 shadow copy for the following WMMA pooling GEMM.
// ---------------------------------------------------------------------------
template<int CIN, int COUT>
__global__ void root_bias_elu_kernel(const float* __restrict__ agg,
                                     const float* __restrict__ xin,
                                     const float* __restrict__ root,
                                     const float* __restrict__ bias,
                                     float* __restrict__ out,
                                     _Float16* __restrict__ out16,
                                     int Nn)
{
    long t = (long)blockIdx.x * blockDim.x + threadIdx.x;
    if (t >= (long)Nn * COUT) return;
    long n = t / COUT; int co = (int)(t % COUT);
    float v = agg[t] + bias[co];
    const float* xr = xin + n * CIN;
    for (int ci = 0; ci < CIN; ++ci)
        v += xr[ci] * root[ci * COUT + co];
    float r = elu1(v);
    out[t] = r;
    if (out16) out16[t] = (_Float16)r;
}

__global__ void f32_to_f16_kernel(const float* __restrict__ in,
                                  _Float16* __restrict__ out, int n)
{
    int t = blockIdx.x * blockDim.x + threadIdx.x;
    if (t < n) out[t] = (_Float16)in[t];
}

// ---------------------------------------------------------------------------
// Batched pooling GEMM on the matrix pipes: O[b] = P[M,K] @ H[b][K,C], f16 in,
// f32 accumulate via v_wmma_f32_16x16x32_f16.
// Fast path (M%16==0, K%64==0, WV==4): 4 waves/block share a 64x16 B tile.
// The tile is staged into LDS in *fragment order* by all 128 threads
// (1 global_load_b128 + 8 ds_store_b16 each), so every consumer wave reads
// its full 256-bit B fragment with 2 ds_load_b128. Two WMMAs per barrier pair.
// Fragment map (16-bit B 32x16): lane ll = 16*(k>=16) + col, element = k&15.
// A layout: lane<16 K in {0..15}, lane>=16 K in {16..31}
//   -> two contiguous global_load_b128 per lane. C/D: m = vgpr + 8*(lane>=16).
// ---------------------------------------------------------------------------
template<int M, int K, int C, int WV>
__global__ void pool_wmma_kernel(const _Float16* __restrict__ P16, // [M,K]
                                 const _Float16* __restrict__ H16, // [B,K,C]
                                 float* __restrict__ O)            // [B,M,C]
{
    constexpr bool FAST = (M % 16 == 0) && (K % 64 == 0) && (WV == 4);
    const int lane = threadIdx.x;           // 0..31
    const int wv   = threadIdx.y;           // 0..WV-1
    const int hi   = lane >> 4;
    const int l15  = lane & 15;
    const int n0   = blockIdx.x * 16;
    const int n    = n0 + l15;
    const int tileM = blockIdx.y * WV + wv;
    const int mrow  = tileM * 16 + l15;
    const size_t hb = (size_t)blockIdx.z * K * C;

    v8f acc = {};
    if constexpr (FAST) {
        __shared__ _Float16 frag[2][32][24];    // fragment-ordered, 48B rows
        const bool active = tileM < (M / 16);
        const _Float16* arow = P16 + (size_t)mrow * K;
        const int t     = wv * 32 + lane;       // 0..127
        const int srow  = t >> 1;               // staged tile row 0..63
        const int scol0 = (t & 1) * 8;          // staged cols 0..7 or 8..15
        const int kb_s  = srow >> 5;            // which 32-k block
        const int kl_s  = srow & 31;
        const int llb   = (kl_s & 16) + scol0;  // 16*(k>=16) + col
        const int el_s  = kl_s & 15;            // element index within fragment

        for (int kk = 0; kk < K; kk += 64) {
            // ---- stage 64x16 B tile in fragment order (all 128 threads) ----
            v8h r = *(const v8h*)(H16 + hb + (size_t)(kk + srow) * C + n0 + scol0);
#pragma unroll
            for (int j = 0; j < 8; ++j)
                frag[kb_s][llb + j][el_s] = r[j];
            __syncthreads();
            if (active) {
#pragma unroll
                for (int kb = 0; kb < 2; ++kb) {
                    v8h alo = *(const v8h*)(arow + kk + kb * 32 + hi * 16);
                    v8h ahi = *(const v8h*)(arow + kk + kb * 32 + 8 + hi * 16);
                    v8h blo = *(const v8h*)&frag[kb][lane][0];
                    v8h bhi = *(const v8h*)&frag[kb][lane][8];
                    acc = __builtin_amdgcn_wmma_f32_16x16x32_f16(
                              false, cat8(alo, ahi), false, cat8(blo, bhi),
                              (short)0, acc, false, false);
                }
            }
            __syncthreads();
        }
        if (!active) return;
    } else {
        for (int kk = 0; kk < K; kk += 32) {
            v16h a, b;
#pragma unroll
            for (int i = 0; i < 16; ++i) {
                int ka = kk + hi * 16 + i;
                a[i] = (mrow < M && ka < K) ? P16[(size_t)mrow * K + ka] : (_Float16)0.f;
                int kb = kk + hi * 16 + i;
                b[i] = (kb < K) ? H16[hb + (size_t)kb * C + n] : (_Float16)0.f;
            }
            acc = __builtin_amdgcn_wmma_f32_16x16x32_f16(
                      false, a, false, b, (short)0, acc, false, false);
        }
    }
    const size_t ob = (size_t)blockIdx.z * M * C;
#pragma unroll
    for (int r = 0; r < 8; ++r) {
        int m = tileM * 16 + r + hi * 8;
        if ((M % 16 == 0) || m < M) O[ob + (size_t)m * C + n] = acc[r];
    }
}

// ---------------------------------------------------------------------------
__global__ void dense_kernel(const float* __restrict__ in, const float* __restrict__ w,
                             const float* __restrict__ b, float* __restrict__ out,
                             int R, int K, int C, int act)
{
    int t = blockIdx.x * blockDim.x + threadIdx.x;
    if (t >= R * C) return;
    int r = t / C, c = t % C;
    float acc = b[c];
    for (int k = 0; k < K; ++k)
        acc += in[r * K + k] * w[k * C + c];
    out[t] = act ? elu1(acc) : acc;
}

__global__ void reparam_kernel(const float* __restrict__ mu, const float* __restrict__ logvar,
                               const float* __restrict__ eps, float* __restrict__ z, int n)
{
    int t = blockIdx.x * blockDim.x + threadIdx.x;
    if (t < n) z[t] = eps[t] * expf(0.5f * logvar[t]) + mu[t];
}

static inline int ceildiv_l(long a, int b) { return (int)((a + b - 1) / b); }

// ---------------------------------------------------------------------------
extern "C" void kernel_launch(void* const* d_in, const int* in_sizes, int n_in,
                              void* d_out, int out_size, void* d_ws, size_t ws_size,
                              hipStream_t stream) {
    (void)in_sizes; (void)n_in; (void)out_size; (void)ws_size;
    const float* x   = (const float*)d_in[0];
    const float* ea0 = (const float*)d_in[1];
    const float* ea1 = (const float*)d_in[2];
    const float* ea2 = (const float*)d_in[3];
    const float* ea3 = (const float*)d_in[4];
    const float* ea4 = (const float*)d_in[5];
    const float* P01 = (const float*)d_in[6];
    const float* P12 = (const float*)d_in[7];
    const float* P23 = (const float*)d_in[8];
    const float* P34 = (const float*)d_in[9];
    const float* Pn1 = (const float*)d_in[10];
    const float* P1n = (const float*)d_in[11];
    const float* P43 = (const float*)d_in[12];
    const float* P32 = (const float*)d_in[13];
    const float* P21 = (const float*)d_in[14];
    const float* P10 = (const float*)d_in[15];
    const float *W_c1=(const float*)d_in[16], *r_c1=(const float*)d_in[17], *b_c1=(const float*)d_in[18];
    const float *W_c2=(const float*)d_in[19], *r_c2=(const float*)d_in[20], *b_c2=(const float*)d_in[21];
    const float *W_c3=(const float*)d_in[22], *r_c3=(const float*)d_in[23], *b_c3=(const float*)d_in[24];
    const float *W_c4=(const float*)d_in[25], *r_c4=(const float*)d_in[26], *b_c4=(const float*)d_in[27];
    const float *W_c5=(const float*)d_in[28], *r_c5=(const float*)d_in[29], *b_c5=(const float*)d_in[30];
    const float *W_d5=(const float*)d_in[31], *r_d5=(const float*)d_in[32], *b_d5=(const float*)d_in[33];
    const float *W_d4=(const float*)d_in[34], *r_d4=(const float*)d_in[35], *b_d4=(const float*)d_in[36];
    const float *W_d3=(const float*)d_in[37], *r_d3=(const float*)d_in[38], *b_d3=(const float*)d_in[39];
    const float *W_d2=(const float*)d_in[40], *r_d2=(const float*)d_in[41], *b_d2=(const float*)d_in[42];
    const float *W_d1=(const float*)d_in[43], *r_d1=(const float*)d_in[44], *b_d1=(const float*)d_in[45];
    const float *fce1_w=(const float*)d_in[46], *fce1_b=(const float*)d_in[47];
    const float *fce21_w=(const float*)d_in[48], *fce21_b=(const float*)d_in[49];
    const float *fce22_w=(const float*)d_in[50], *fce22_b=(const float*)d_in[51];
    const float *fcd3_w=(const float*)d_in[52], *fcd3_b=(const float*)d_in[53];
    const float *fcd4_w=(const float*)d_in[54], *fcd4_b=(const float*)d_in[55];
    const float *eps = (const float*)d_in[56];
    const int *ei0=(const int*)d_in[57], *ei1=(const int*)d_in[58], *ei2=(const int*)d_in[59],
              *ei3=(const int*)d_in[60], *ei4=(const int*)d_in[61];

    float* outp = (float*)d_out;           // [recon 65536][mu 512][logvar 512]
    float* mu_o = outp + 65536;
    float* lv_o = outp + 66048;

    // ---- workspace layout ----
    float* buf0 = (float*)d_ws;            // f32 feature ping-pong (1M floats each)
    float* buf1 = buf0 + (1 << 20);
    float* buf2 = buf1 + (1 << 20);        // scatter accumulator
    float* s0   = buf2 + (1 << 20);        // small f32 slots (4096 each)
    float* s1   = s0 + 4096;
    float* s2   = s1 + 4096;
    _Float16* h16 = (_Float16*)(s2 + 4096);    // f16 shadow of latest conv output
    _Float16* s16 = h16 + (1 << 20);           // f16 shadow of dense output
    _Float16* p01h = s16 + 4096;               // f16 copies of pooling matrices
    _Float16* p12h = p01h + 1024L * 4096;
    _Float16* p23h = p12h + 256L * 1024;
    _Float16* p34h = p23h + 64L * 256;
    _Float16* pn1h = p34h + 16L * 64;
    _Float16* p1nh = pn1h + 16;
    _Float16* p43h = p1nh + 16;
    _Float16* p32h = p43h + 64L * 16;
    _Float16* p21h = p32h + 256L * 64;
    _Float16* p10h = p21h + 1024L * 256;

    const int E0 = 16 * 4096 * 6, E1 = 16 * 1024 * 6, E2 = 16 * 256 * 6,
              E3 = 16 * 64 * 6,   E4 = 16 * 16 * 6;

#define CVT(SRC, DST, N)                                                                         \
    f32_to_f16_kernel<<<ceildiv_l((long)(N), 256), 256, 0, stream>>>(SRC, DST, (int)(N));

    // convert pooling matrices to f16 once (each reused by 16 batched GEMMs)
    CVT(P01, p01h, 1024L * 4096);  CVT(P12, p12h, 256L * 1024);
    CVT(P23, p23h, 64L * 256);     CVT(P34, p34h, 16L * 64);
    CVT(Pn1, pn1h, 16);            CVT(P1n, p1nh, 16);
    CVT(P43, p43h, 64L * 16);      CVT(P32, p32h, 256L * 64);
    CVT(P21, p21h, 1024L * 256);   CVT(P10, p10h, 1024L * 4096);

#define CONV(CIN, COUT, XIN, EIDX, EA, WW, ROOT, BIAS, OUT, OUT16, NN, EE)                       \
    hipMemsetAsync(buf2, 0, (size_t)(NN) * (COUT) * sizeof(float), stream);                      \
    spline_msg_kernel<CIN, COUT><<<ceildiv_l((long)(EE) * (COUT), 256), 256, 0, stream>>>(       \
        XIN, EIDX, EA, WW, buf2, EE);                                                            \
    root_bias_elu_kernel<CIN, COUT><<<ceildiv_l((long)(NN) * (COUT), 256), 256, 0, stream>>>(    \
        buf2, XIN, ROOT, BIAS, OUT, OUT16, NN);

#define POOL4(M, K, C, PM, HIN, OOUT)                                                            \
    pool_wmma_kernel<M, K, C, 4><<<dim3((C) / 16, ((M) / 16 + 3) / 4, 16),                       \
                                   dim3(32, 4), 0, stream>>>(PM, HIN, OOUT);
#define POOL1(M, K, C, PM, HIN, OOUT)                                                            \
    pool_wmma_kernel<M, K, C, 1><<<dim3((C) / 16, ((M) + 15) / 16, 16),                          \
                                   dim3(32, 1), 0, stream>>>(PM, HIN, OOUT);

    // ----------------- encoder -----------------
    CONV(1, 16, x, ei0, ea0, W_c1, r_c1, b_c1, buf0, h16, 65536, E0);
    POOL4(1024, 4096, 16, p01h, h16, buf1);
    CONV(16, 32, buf1, ei1, ea1, W_c2, r_c2, b_c2, buf0, h16, 16384, E1);
    POOL4(256, 1024, 32, p12h, h16, buf1);
    CONV(32, 64, buf1, ei2, ea2, W_c3, r_c3, b_c3, buf0, h16, 4096, E2);
    POOL4(64, 256, 64, p23h, h16, buf1);
    CONV(64, 128, buf1, ei3, ea3, W_c4, r_c4, b_c4, buf0, h16, 1024, E3);
    POOL4(16, 64, 128, p34h, h16, buf1);
    CONV(128, 256, buf1, ei4, ea4, W_c5, r_c5, b_c5, buf0, h16, 256, E4);
    POOL1(1, 16, 256, pn1h, h16, buf1);                // h5: [16,256]

    dense_kernel<<<16, 256, 0, stream>>>(buf1, fce1_w, fce1_b, s0, 16, 256, 256, 1);
    dense_kernel<<<2, 256, 0, stream>>>(s0, fce21_w, fce21_b, mu_o, 16, 256, 32, 0);
    dense_kernel<<<2, 256, 0, stream>>>(s0, fce22_w, fce22_b, lv_o, 16, 256, 32, 0);
    reparam_kernel<<<2, 256, 0, stream>>>(mu_o, lv_o, eps, s1, 512);

    // ----------------- decoder -----------------
    dense_kernel<<<16, 256, 0, stream>>>(s1, fcd3_w, fcd3_b, s2, 16, 32, 256, 1);
    dense_kernel<<<16, 256, 0, stream>>>(s2, fcd4_w, fcd4_b, s0, 16, 256, 256, 1);
    CVT(s0, s16, 4096);
    POOL1(16, 1, 256, p1nh, s16, buf0);                // [16,16,256] = 256 nodes
    CONV(256, 128, buf0, ei4, ea4, W_d5, r_d5, b_d5, buf1, h16, 256, E4);
    POOL1(64, 16, 128, p43h, h16, buf0);
    CONV(128, 64, buf0, ei3, ea3, W_d4, r_d4, b_d4, buf1, h16, 1024, E3);
    POOL4(256, 64, 64, p32h, h16, buf0);
    CONV(64, 32, buf0, ei2, ea2, W_d3, r_d3, b_d3, buf1, h16, 4096, E2);
    POOL4(1024, 256, 32, p21h, h16, buf0);
    CONV(32, 16, buf0, ei1, ea1, W_d2, r_d2, b_d2, buf1, h16, 16384, E1);
    POOL4(4096, 1024, 16, p10h, h16, buf0);
    CONV(16, 1, buf0, ei0, ea0, W_d1, r_d1, b_d1, outp, (_Float16*)nullptr, 65536, E0);

#undef CVT
#undef CONV
#undef POOL4
#undef POOL1
}
